// MLP_28295244546755
// MI455X (gfx1250) — matrix operand order repaired and checked by reading, not compiled, not run on hardware
//
#include <hip/hip_runtime.h>
#include <hip/hip_bf16.h>

// ---------------------------------------------------------------------------
// MLP encode -> L2 normalize -> dense cosine sim -> top-(k+1) mask -> relu
// N=8192, D=256, k=30.
// Traffic plan (HBM-bound problem): the only large stream is one 268MB zero
// fill of d_out.  sim is NEVER materialized: each 16-row block sweeps all
// 8192 columns with bf16 WMMA, maintains exact per-row top-31 in LDS, and
// scatters <=31 relu(sim) values per row at the end.
// ---------------------------------------------------------------------------

typedef __attribute__((ext_vector_type(16))) __bf16 v16bf;
typedef __attribute__((ext_vector_type(8)))  __bf16 v8bf;
typedef __attribute__((ext_vector_type(8)))  float  v8f;
typedef __attribute__((ext_vector_type(4)))  float  v4f;

#define NROWS 8192
#define DDIM  256
#define KCAP  31
#define NEG_INF (-3.402823e38f)

__device__ __forceinline__ v8f wmma_bf16(v16bf a, v16bf b, v8f c) {
    // (neg_a, A, neg_b, B, c_mod, C, reuse_a, reuse_b)
    return __builtin_amdgcn_wmma_f32_16x16x32_bf16(false, a, false, b,
                                                   (short)0, c, false, false);
}

// A-operand fragment (16x32 bf16): lane L holds row
// M=L&15, 16 contiguous K-values at (L>>4)*16 (matches B operand).
__device__ __forceinline__ v16bf load_a_bf16(const __bf16* row_base, int c0) {
    v8bf lo = *(const v8bf*)(row_base + c0);
    v8bf hi = *(const v8bf*)(row_base + c0 + 8);
    v16bf a;
#pragma unroll
    for (int i = 0; i < 8; ++i) { a[i] = lo[i]; a[8 + i] = hi[i]; }
    return a;
}

__device__ __forceinline__ v16bf load_a_f32(const float* row_base, int c0) {
    v8f lo = *(const v8f*)(row_base + c0);
    v8f hi = *(const v8f*)(row_base + c0 + 8);
    v16bf a;
#pragma unroll
    for (int i = 0; i < 8; ++i) { a[i] = (__bf16)lo[i]; a[8 + i] = (__bf16)hi[i]; }
    return a;
}

// B-operand fragment (32x16 bf16, KxN): lane L holds column N=L&15 as one
// contiguous chunk of 16 K-values at (L>>4)*16.  B column n == source row n.
__device__ __forceinline__ v16bf load_b_f32(const float* row_base, int c0) {
    v8f lo = *(const v8f*)(row_base + c0);
    v8f hi = *(const v8f*)(row_base + c0 + 8);
    v16bf b;
#pragma unroll
    for (int i = 0; i < 8; ++i) { b[i] = (__bf16)lo[i]; b[8 + i] = (__bf16)hi[i]; }
    return b;
}

// ---------------------------------------------------------------------------
// Kernel 0: stream zeros into d_out (the single large HBM stream).
// ---------------------------------------------------------------------------
__global__ __launch_bounds__(256) void k_zero(float* __restrict__ out) {
    const size_t n4 = (size_t)NROWS * NROWS / 4;
    size_t i = (size_t)blockIdx.x * blockDim.x + threadIdx.x;
    const size_t stride = (size_t)gridDim.x * blockDim.x;
    v4f z = {};
    for (; i < n4; i += stride)
        __builtin_nontemporal_store(z, (v4f*)out + i);
}

// ---------------------------------------------------------------------------
// Kernel 1: fused  h = relu(X@W1^T + b1);  emb = h@W2^T + b2;  rows L2-norm;
// store normalized emb as bf16.  One block = 16 rows, 8 waves, each wave owns
// two 16-col tiles of the 256-wide output.  Hidden tile lives in LDS only.
// ---------------------------------------------------------------------------
__global__ __launch_bounds__(256) void k_mlp(const float* __restrict__ X,
                                             const float* __restrict__ W1,
                                             const float* __restrict__ b1,
                                             const float* __restrict__ W2,
                                             const float* __restrict__ b2,
                                             __bf16* __restrict__ embN) {
    __shared__ __bf16 hsh[16 * DDIM];     // 8KB hidden tile
    __shared__ float  part[16][16];       // per-row, per-tile |emb|^2 partials
    __shared__ float  rowinv[16];

    const int tid  = threadIdx.x;
    const int wave = tid >> 5;
    const int lane = tid & 31;
    const int lm   = lane & 15;
    const int lh   = lane >> 4;
    const int rbase = blockIdx.x * 16;

    v16bf afrag[8];
#pragma unroll
    for (int kk = 0; kk < 8; ++kk)
        afrag[kk] = load_a_f32(X + (size_t)(rbase + lm) * DDIM + kk * 32, lh * 16);

#pragma unroll
    for (int t = 0; t < 2; ++t) {
        const int nb = wave * 32 + t * 16;
        v8f c = {};
#pragma unroll
        for (int kk = 0; kk < 8; ++kk) {
            v16bf b = load_b_f32(W1 + (size_t)(nb + lm) * DDIM + kk * 32, lh * 16);
            c = wmma_bf16(afrag[kk], b, c);
        }
        const float bias = b1[nb + lm];
#pragma unroll
        for (int j = 0; j < 8; ++j) {
            float v = fmaxf(c[j] + bias, 0.0f);
            hsh[(j + lh * 8) * DDIM + nb + lm] = (__bf16)v;
        }
    }
    __syncthreads();

    v16bf hfrag[8];
#pragma unroll
    for (int kk = 0; kk < 8; ++kk)
        hfrag[kk] = load_a_bf16(hsh + lm * DDIM + kk * 32, lh * 16);

    v8f cc[2];
#pragma unroll
    for (int t = 0; t < 2; ++t) {
        const int nb = wave * 32 + t * 16;
        v8f c = {};
#pragma unroll
        for (int kk = 0; kk < 8; ++kk) {
            v16bf b = load_b_f32(W2 + (size_t)(nb + lm) * DDIM + kk * 32, lh * 16);
            c = wmma_bf16(hfrag[kk], b, c);
        }
        const float bias = b2[nb + lm];
#pragma unroll
        for (int j = 0; j < 8; ++j) c[j] += bias;
        cc[t] = c;
        // deterministic |row|^2 partials: reduce across each 16-lane group
#pragma unroll
        for (int j = 0; j < 8; ++j) {
            float s = c[j] * c[j];
            s += __shfl_xor(s, 1);
            s += __shfl_xor(s, 2);
            s += __shfl_xor(s, 4);
            s += __shfl_xor(s, 8);
            if (lm == 0) part[j + lh * 8][wave * 2 + t] = s;
        }
    }
    __syncthreads();
    if (tid < 16) {
        float s = 0.0f;
#pragma unroll
        for (int i = 0; i < 16; ++i) s += part[tid][i];
        rowinv[tid] = 1.0f / fmaxf(sqrtf(s), 1e-12f);
    }
    __syncthreads();

#pragma unroll
    for (int t = 0; t < 2; ++t) {
        const int nb = wave * 32 + t * 16;
#pragma unroll
        for (int j = 0; j < 8; ++j) {
            const int rl = j + lh * 8;
            float v = cc[t][j] * rowinv[rl];
            embN[(size_t)(rbase + rl) * DDIM + nb + lm] = (__bf16)v;
        }
    }
}

// ---------------------------------------------------------------------------
// Kernel 2: fused sim sweep + exact per-row top-(k+1) + scatter.
// Block = 16 rows.  64 batches of 128 columns; per batch each of 8 waves does
// 8 WMMAs and stages its 16x16 C tile in LDS (double buffered, padded stride
// 132 -> conflict-free).  128 selector threads (8 per row) keep exact
// per-thread top-31 lists (stride-33 LDS, running min in registers).
// Cooperative merge then <=31 scatter stores per row (diag excluded).
// ---------------------------------------------------------------------------
#define STG_LD 132              // padded stage row stride (floats)
#define LST_LD 33               // padded list stride (entries)

__global__ __launch_bounds__(256) void k_simtopk(const __bf16* __restrict__ embN,
                                                 float* __restrict__ out,
                                                 const int* __restrict__ kptr) {
    __shared__ float stage[2 * 16 * STG_LD];   // ~16.9KB
    __shared__ float listv[128 * LST_LD];      // ~16.9KB
    __shared__ int   listi[128 * LST_LD];      // ~16.9KB
    __shared__ float cand[128];
    __shared__ int   candp[128];
    __shared__ float selv[16][KCAP];
    __shared__ int   seli[16][KCAP];

    const int t    = threadIdx.x;
    const int wave = t >> 5;
    const int lane = t & 31;
    const int lm   = lane & 15;
    const int lh   = lane >> 4;
    const int rbase = blockIdx.x * 16;

    int kin = kptr[0] + 1;                    // top-(k+1), diag removed later
    const int ksel = kin < KCAP ? kin : KCAP;

    // init per-thread candidate lists (self-owned until merge)
    if (t < 128) {
#pragma unroll
        for (int i = 0; i < KCAP; ++i) {
            listv[t * LST_LD + i] = NEG_INF;
            listi[t * LST_LD + i] = 0;
        }
    }
    float curminv = NEG_INF;
    int   curminp = 0;

    // A fragments for this block's 16 rows (held in VGPRs for whole sweep)
    v16bf afrag[8];
#pragma unroll
    for (int kk = 0; kk < 8; ++kk)
        afrag[kk] = load_a_bf16(embN + (size_t)(rbase + lm) * DDIM + kk * 32, lh * 16);

    const int selr = t >> 3;                  // selector: row 0..15
    const int selq = t & 7;                   //           slot 0..7

    for (int bt = 0; bt < NROWS / 128; ++bt) {
        const int cbase0 = bt * 128;
        const int cbase  = cbase0 + wave * 16;
        const int p      = bt & 1;

        if (bt + 1 < NROWS / 128)             // hint next batch of B rows
            __builtin_prefetch(embN + (size_t)(cbase + 128 + lm) * DDIM, 0, 1);

        v8f c = {};
#pragma unroll
        for (int kk = 0; kk < 8; ++kk) {
            v16bf b = *(const v16bf*)(embN + (size_t)(cbase + lm) * DDIM +
                                      kk * 32 + lh * 16);
            c = wmma_bf16(afrag[kk], b, c);
        }
        // stage C tile: stage[p][row][wave*16+lm]
        float* sp = stage + p * 16 * STG_LD;
#pragma unroll
        for (int j = 0; j < 8; ++j)
            sp[(j + lh * 8) * STG_LD + wave * 16 + lm] = c[j];

        __syncthreads();   // double buffer: scans of batch bt overlap next stores

        if (t < 128) {     // 8 selector threads per row scan 16 cols each
            const float* sr = stage + p * 16 * STG_LD + selr * STG_LD + selq * 16;
            for (int i = 0; i < 16; ++i) {
                const float v = sr[i];
                if (v > curminv) {
                    listv[t * LST_LD + curminp] = v;
                    listi[t * LST_LD + curminp] = cbase0 + selq * 16 + i;
                    float nm = listv[t * LST_LD];
                    int   np = 0;
                    for (int z = 1; z < KCAP; ++z) {
                        const float lv = listv[t * LST_LD + z];
                        if (lv < nm) { nm = lv; np = z; }
                    }
                    curminv = nm; curminp = np;
                }
            }
        }
    }
    __syncthreads();

    // cooperative exact merge: per row, 8 lists of 31 -> global top-ksel
    for (int s = 0; s < ksel; ++s) {
        if (t < 128) {
            float mv = NEG_INF; int mp = 0;
            for (int i = 0; i < KCAP; ++i) {
                const float v = listv[t * LST_LD + i];
                if (v > mv) { mv = v; mp = i; }
            }
            cand[t] = mv; candp[t] = mp;
        }
        __syncthreads();
        if (t < 16) {
            float bv = NEG_INF; int bq = 0;
#pragma unroll
            for (int q = 0; q < 8; ++q) {
                const float v = cand[t * 8 + q];
                if (v > bv) { bv = v; bq = q; }
            }
            const int w = t * 8 + bq;
            selv[t][s] = bv;
            seli[t][s] = listi[w * LST_LD + candp[w]];
            listv[w * LST_LD + candp[w]] = NEG_INF;   // remove winner
        }
        __syncthreads();
    }

    // scatter: out already zero-filled; write relu(sim) at selected cols,
    // skipping the diagonal (fill_diagonal_(0)).
    for (int e = t; e < 16 * ksel; e += 256) {
        const int r = e / ksel;
        const int s = e - r * ksel;
        const int col  = seli[r][s];
        const int grow = rbase + r;
        if (col != grow)
            out[(size_t)grow * NROWS + col] = fmaxf(selv[r][s], 0.0f);
    }
}

// ---------------------------------------------------------------------------
extern "C" void kernel_launch(void* const* d_in, const int* in_sizes, int n_in,
                              void* d_out, int out_size, void* d_ws, size_t ws_size,
                              hipStream_t stream) {
    const float* X  = (const float*)d_in[0];
    const float* W1 = (const float*)d_in[1];
    const float* b1 = (const float*)d_in[2];
    const float* W2 = (const float*)d_in[3];
    const float* b2 = (const float*)d_in[4];
    const int*   kp = (const int*)d_in[5];
    float* out = (float*)d_out;
    __bf16* embN = (__bf16*)d_ws;           // 8192*256*2 = 4MB scratch

    k_zero   <<<2048,       256, 0, stream>>>(out);
    k_mlp    <<<NROWS / 16, 256, 0, stream>>>(X, W1, b1, W2, b2, embN);
    k_simtopk<<<NROWS / 16, 256, 0, stream>>>(embN, out, kp);
}
